// Word2VectorNegSampling_20023137534888
// MI455X (gfx1250) — matrix operand order repaired and checked by reading, not compiled, not run on hardware
//
#include <hip/hip_runtime.h>
#include <math.h>

// Problem constants (from reference)
#define B_DIM 2048
#define V_DIM 100000
#define D_DIM 128
#define K_CH  21

// GEMM tiling
#define BM 128
#define BN 128
#define BK 32
#define SPLITK 64
#define KSLICE 1568          // ceil(100000/64) rounded up to BK; 64*1568 >= 100000
#define LDA 40               // LDS row stride in ushort (80B = 20 banks -> conflict free)
#define LDB 40

typedef __attribute__((ext_vector_type(16))) __bf16        v16bf;
typedef __attribute__((ext_vector_type(8)))  float         v8f;
typedef __attribute__((ext_vector_type(4)))  float         v4f;
typedef __attribute__((ext_vector_type(4)))  unsigned int  v4u;
typedef __attribute__((ext_vector_type(8)))  unsigned int  v8u;
typedef __attribute__((ext_vector_type(2)))  unsigned int  v2u;

// ---- helpers ---------------------------------------------------------------

// pack two f32 into one dword of two bf16 (RNE) with the CDNA5 hw instruction
__device__ __forceinline__ unsigned int cvt_pk_bf16(float a, float b) {
    unsigned int r;
    asm("v_cvt_pk_bf16_f32 %0, %1, %2" : "=v"(r) : "v"(a), "v"(b));
    return r;
}

// store 4 f32 as 4 packed bf16 (8 bytes, one ds_store_b64)
__device__ __forceinline__ void store_bf16x4(unsigned short* dst, v4f v) {
    v2u p = {cvt_pk_bf16(v.x, v.y), cvt_pk_bf16(v.z, v.w)};
    *(v2u*)dst = p;
}

// build a v16bf fragment from two 16B LDS reads (ds_load_b128 x2)
__device__ __forceinline__ v16bf ld_frag(const unsigned short* p0,
                                         const unsigned short* p1) {
    v4u a = *(const v4u*)p0;
    v4u b = *(const v4u*)p1;
    v8u c = {a.x, a.y, a.z, a.w, b.x, b.y, b.z, b.w};
    return __builtin_bit_cast(v16bf, c);
}

// ---- kernel 1: split-K GEMM  x_partial[s] = inputs @ W_emb^T (K-slice) -----
// inputs [B,V] f32 (streamed, non-temporal), W_emb [D,V] f32 (L2 resident),
// partials [SPLITK][B][D] f32.  Double-buffered LDS, one barrier per K-step.
__global__ __launch_bounds__(256)
void gemm_bf16_splitk(const float* __restrict__ A,
                      const float* __restrict__ Wemb,
                      float* __restrict__ partials) {
    __shared__ unsigned short As[2][BM * LDA];
    __shared__ unsigned short Bs[2][BN * LDB];

    const int m0    = blockIdx.x * BM;       // 16 M-blocks
    const int slice = blockIdx.y;            // 64 K-slices
    const int kbeg  = slice * KSLICE;
    const int kend  = (kbeg + KSLICE < V_DIM) ? (kbeg + KSLICE) : V_DIM;

    const int tid  = threadIdx.x;
    const int lane = tid & 31;
    const int wave = tid >> 5;               // 8 waves
    const int wm   = wave >> 1;              // 0..3 -> rows 32*wm
    const int wn   = wave & 1;               // 0..1 -> cols 64*wn
    const int l16  = lane & 15;
    const int half = lane >> 4;

    // staging: thread owns rows r0+32*it (it<4) at cols [c4, c4+4)
    const int r0 = tid >> 3;                 // 0..31
    const int c4 = (tid & 7) << 2;           // 0,4,..,28
    const float* pa = A    + (size_t)(m0 + r0) * V_DIM + kbeg + c4;
    const float* pb = Wemb + (size_t)r0       * V_DIM + kbeg + c4;
    const size_t rstride = (size_t)32 * V_DIM;

    v8f acc[2][4];
#pragma unroll
    for (int i = 0; i < 2; ++i)
#pragma unroll
        for (int j = 0; j < 4; ++j) acc[i][j] = (v8f)0.0f;

    // prologue: fetch first tile into regs (tiles are always full: V%32==0)
    v4f ra[4], rb[4];
#pragma unroll
    for (int it = 0; it < 4; ++it) {
        ra[it] = __builtin_nontemporal_load((const v4f*)(pa + it * rstride));
        rb[it] = *(const v4f*)(pb + it * rstride);
    }

    int buf = 0;
    for (int kb = kbeg; kb < kend; kb += BK) {
        // ---- store current tile regs -> LDS[buf] as bf16 ------------------
#pragma unroll
        for (int it = 0; it < 4; ++it) {
            int row = r0 + 32 * it;
            store_bf16x4(&As[buf][row * LDA + c4], ra[it]);
            store_bf16x4(&Bs[buf][row * LDB + c4], rb[it]);
        }
        __syncthreads();

        // ---- issue next tile's global loads (overlap with WMMA) -----------
        if (kb + BK < kend) {                // uniform branch
            pa += BK; pb += BK;
#pragma unroll
            for (int it = 0; it < 4; ++it) {
                ra[it] = __builtin_nontemporal_load((const v4f*)(pa + it * rstride));
                rb[it] = *(const v4f*)(pb + it * rstride);
            }
        }

        // ---- fragments + 8 WMMAs per wave ---------------------------------
        v16bf af[2], bfm[4];
#pragma unroll
        for (int i = 0; i < 2; ++i) {
            const unsigned short* p =
                &As[buf][(wm * 32 + 16 * i + l16) * LDA + 8 * half];
            af[i] = ld_frag(p, p + 16);      // K runs {0-7,16-23} / {8-15,24-31}
        }
#pragma unroll
        for (int j = 0; j < 4; ++j) {
            const unsigned short* p =
                &Bs[buf][(wn * 64 + 16 * j + l16) * LDB + 8 * half];
            bfm[j] = ld_frag(p, p + 16);     // K runs {0-7,16-23} / {8-15,24-31}
        }
#pragma unroll
        for (int i = 0; i < 2; ++i)
#pragma unroll
            for (int j = 0; j < 4; ++j)
                acc[i][j] = __builtin_amdgcn_wmma_f32_16x16x32_bf16(
                    false, af[i], false, bfm[j], (short)0, acc[i][j],
                    false, false);
        buf ^= 1;
    }

    // ---- write f32 partial tile ------------------------------------------
    float* outp = partials + (size_t)slice * (B_DIM * D_DIM);
#pragma unroll
    for (int i = 0; i < 2; ++i)
#pragma unroll
        for (int j = 0; j < 4; ++j) {
            int n = wn * 64 + 16 * j + l16;
#pragma unroll
            for (int r = 0; r < 8; ++r) {
                int m = m0 + wm * 32 + 16 * i + half * 8 + r;
                outp[(size_t)m * D_DIM + n] = acc[i][j][r];
            }
        }
}

// ---- kernel 2: reduce split-K partials -> x [B,D] (float4 per thread) ------
__global__ __launch_bounds__(256)
void reduce_partials(const float* __restrict__ partials,
                     float* __restrict__ x) {
    int i = (blockIdx.x * 256 + threadIdx.x) * 4;  // 0 .. B*D-1 step 4
    v4f s = (v4f)0.0f;
#pragma unroll 8
    for (int p = 0; p < SPLITK; ++p) {
        v4f v = *(const v4f*)&partials[(size_t)p * (B_DIM * D_DIM) + i];
        s.x += v.x; s.y += v.y; s.z += v.z; s.w += v.w;
    }
    *(v4f*)&x[i] = s;
}

// ---- kernel 3: gathered logits + per-row BCE partial -----------------------
// one wave per batch row b; lanes parallel over D (4 f32 each)
__global__ __launch_bounds__(256)
void loss_partial(const float* __restrict__ x,
                  const float* __restrict__ Wcls,
                  const int* __restrict__ ids,
                  const int* __restrict__ labels,
                  float* __restrict__ lpart) {
    const int wave = threadIdx.x >> 5;
    const int lane = threadIdx.x & 31;
    const int b = blockIdx.x * 8 + wave;

    v4f xv = *(const v4f*)&x[(size_t)b * D_DIM + lane * 4];
    float acc = 0.f;
    for (int k = 0; k < K_CH; ++k) {
        int id = ids[b * K_CH + k];
        v4f wv = *(const v4f*)&Wcls[(size_t)id * D_DIM + lane * 4];
        float p = xv.x * wv.x + xv.y * wv.y + xv.z * wv.z + xv.w * wv.w;
#pragma unroll
        for (int off = 16; off > 0; off >>= 1) p += __shfl_xor(p, off, 32);
        if (lane == 0) {
            float t = (float)labels[b * K_CH + k];
            acc += fmaxf(p, 0.f) - p * t + log1pf(expf(-fabsf(p)));
        }
    }
    if (lane == 0) lpart[b] = acc;
}

// ---- kernel 4: deterministic final mean ------------------------------------
__global__ __launch_bounds__(256)
void final_reduce(const float* __restrict__ lpart, float* __restrict__ out) {
    __shared__ float sm[256];
    float s = 0.f;
    for (int i = threadIdx.x; i < B_DIM; i += 256) s += lpart[i];
    sm[threadIdx.x] = s;
    __syncthreads();
    for (int st = 128; st > 0; st >>= 1) {
        if (threadIdx.x < st) sm[threadIdx.x] += sm[threadIdx.x + st];
        __syncthreads();
    }
    if (threadIdx.x == 0) out[0] = sm[0] * (1.0f / (float)(B_DIM * K_CH));
}

// ---- launch ----------------------------------------------------------------
extern "C" void kernel_launch(void* const* d_in, const int* in_sizes, int n_in,
                              void* d_out, int out_size, void* d_ws, size_t ws_size,
                              hipStream_t stream) {
    const float* inputs = (const float*)d_in[0];   // [B,V]
    const int*   ids    = (const int*)d_in[1];     // [B,K]
    const int*   labels = (const int*)d_in[2];     // [B,K]
    const float* Wemb   = (const float*)d_in[3];   // [D,V]
    const float* Wcls   = (const float*)d_in[4];   // [V,D]
    float* out = (float*)d_out;

    char* ws = (char*)d_ws;
    const size_t part_bytes = (size_t)SPLITK * B_DIM * D_DIM * sizeof(float); // 64 MB
    const size_t x_bytes    = (size_t)B_DIM * D_DIM * sizeof(float);          // 1 MB
    float* partials = (float*)ws;
    float* x        = (float*)(ws + part_bytes);
    float* lpart    = (float*)(ws + part_bytes + x_bytes);

    gemm_bf16_splitk<<<dim3(B_DIM / BM, SPLITK), 256, 0, stream>>>(inputs, Wemb, partials);
    reduce_partials<<<dim3((B_DIM * D_DIM) / 1024), 256, 0, stream>>>(partials, x);
    loss_partial<<<dim3(B_DIM / 8), 256, 0, stream>>>(x, Wcls, ids, labels, lpart);
    final_reduce<<<1, 256, 0, stream>>>(lpart, out);
}
